// ViTtri_48567490183358
// MI455X (gfx1250) — matrix operand order repaired and checked by reading, not compiled, not run on hardware
//
#include <hip/hip_runtime.h>
#include <hip/hip_bf16.h>
#include <cstdint>

typedef __attribute__((ext_vector_type(16))) _Float16 v16h;
typedef __attribute__((ext_vector_type(8)))  _Float16 v8h;
typedef __attribute__((ext_vector_type(8)))  float    v8f;

#define WMMA_F32_F16(Aop, Bop, Cop) \
  __builtin_amdgcn_wmma_f32_16x16x32_f16(false, (Aop), false, (Bop), (short)0, (Cop), false, false)

// ---------------- model constants ----------------
namespace vt {
constexpr int Bc   = 2;
constexpr int Tt   = 197;   // tokens (196 patches + cls)
constexpr int DIM  = 256;
constexpr int H    = 4;
constexpr int DH   = 64;
constexpr int MLP  = 1024;
constexpr int NCLS = 1000;
constexpr int NP   = 196;
constexpr int PD   = 768;
constexpr int GR   = 14;
constexpr int NQT  = 13;    // ceil(197/16) query tiles
constexpr int NTT  = 13;    // ceil(197/16) key (t) tiles
constexpr int SW   = 25;    // ceil(197/8) s-strip per wave
constexpr int TP   = 208;   // padded token count for transposed e (multiple of 16)
}

// ---------------- helpers ----------------
__device__ __forceinline__ float block_reduce_sum_256(float v, float* sbuf) {
  int tid = threadIdx.x;
  sbuf[tid] = v;
  __syncthreads();
  for (int s = 128; s > 0; s >>= 1) {
    if (tid < s) sbuf[tid] += sbuf[tid + s];
    __syncthreads();
  }
  float r = sbuf[0];
  __syncthreads();
  return r;
}

// ---------------- zero-fill f16 ----------------
__global__ void fill_zero_f16_kernel(_Float16* __restrict__ y, int n) {
  for (int i = blockIdx.x * blockDim.x + threadIdx.x; i < n; i += gridDim.x * blockDim.x)
    y[i] = (_Float16)0.f;
}

// ---------------- generic cast f32 -> f16 ----------------
__global__ void cast_f32_f16_kernel(const float* __restrict__ x, _Float16* __restrict__ y, int n) {
  for (int i = blockIdx.x * blockDim.x + threadIdx.x; i < n; i += gridDim.x * blockDim.x)
    y[i] = (_Float16)x[i];
}

// ---------------- cast + transpose: W[K,N] f32 -> Wt[N,K] f16 (K-contiguous per column) ----------------
__global__ void cast_transpose_f16_kernel(const float* __restrict__ W, _Float16* __restrict__ Wt,
                                          int K, int N) {
  int n = K * N;
  for (int i = blockIdx.x * blockDim.x + threadIdx.x; i < n; i += gridDim.x * blockDim.x) {
    int k = i / N, c = i % N;                // coalesced read along N
    Wt[(size_t)c * K + k] = (_Float16)W[i];
  }
}

// ---------------- patchify + LN1 -> f16 rows [B*196, 768] ----------------
__global__ __launch_bounds__(256) void patchify_ln_kernel(
    const float* __restrict__ img, const float* __restrict__ g, const float* __restrict__ b,
    _Float16* __restrict__ out) {
  __shared__ float sbuf[256];
  int row  = blockIdx.x;            // b*196 + patch
  int bi   = row / vt::NP;
  int pidx = row % vt::NP;
  int gh = pidx / vt::GR, gw = pidx % vt::GR;
  int tid = threadIdx.x;
  float vals[3];
  float s = 0.f;
#pragma unroll
  for (int it = 0; it < 3; ++it) {
    int j = tid + it * 256;         // (p1*16+p2)*3 + c
    int c = j % 3, rem = j / 3, p2 = rem % 16, p1 = rem / 16;
    float v = img[(((size_t)bi * 3 + c) * 224 + gh * 16 + p1) * 224 + gw * 16 + p2];
    vals[it] = v;
    s += v;
  }
  float mean = block_reduce_sum_256(s, sbuf) * (1.f / 768.f);
  float vs = 0.f;
#pragma unroll
  for (int it = 0; it < 3; ++it) { float d = vals[it] - mean; vs += d * d; }
  float var  = block_reduce_sum_256(vs, sbuf) * (1.f / 768.f);
  float rinv = rsqrtf(var + 1e-5f);
#pragma unroll
  for (int it = 0; it < 3; ++it) {
    int j = tid + it * 256;
    out[(size_t)row * vt::PD + j] = (_Float16)((vals[it] - mean) * rinv * g[j] + b[j]);
  }
}

// ---------------- generic WMMA GEMM: C[M,N] = f16A[M,K] x f16Bt[N,K]^T (+bias,+gelu,+addend) ----------------
// Bt is the K-major (transposed) weight so both operand fragments are contiguous vector loads.
__global__ __launch_bounds__(256) void gemm_wmma_kernel(
    const _Float16* __restrict__ A, int lda,
    const _Float16* __restrict__ Bt,
    const float* __restrict__ bias, const float* __restrict__ addend,
    float* __restrict__ C, int ldc,
    int M, int N, int K, int act) {
  int wave = threadIdx.x >> 5;
  int lane = threadIdx.x & 31;
  int l16 = lane & 15, hf = lane >> 4;
  int ntn = (N + 15) >> 4, ntm = (M + 15) >> 4;
  int tile = blockIdx.x * 8 + wave;
  if (tile >= ntm * ntn) return;
  int row0 = (tile / ntn) * 16, col0 = (tile % ntn) * 16;
  int arow = row0 + l16; if (arow >= M) arow = M - 1;
  int bcol = col0 + l16; if (bcol >= N) bcol = N - 1;
  const _Float16* Ar = A + (size_t)arow * lda;
  const _Float16* Br = Bt + (size_t)bcol * K;
  v8f acc{};
  for (int k0 = 0; k0 < K; k0 += 32) {
    __builtin_prefetch(Ar + k0 + 128, 0, 0);   // global_prefetch_b8
    __builtin_prefetch(Br + k0 + 128, 0, 0);
    // A fragment (16x32, f16): elements 0..15 <-> K = k0 + 16*hf + 0..15 (contiguous)
    v8h c0 = *(const v8h*)(Ar + k0 + 16 * hf);
    v8h c1 = *(const v8h*)(Ar + k0 + 16 * hf + 8);
    v16h af;
#pragma unroll
    for (int j = 0; j < 8; j++) { af[j] = c0[j]; af[8 + j] = c1[j]; }
    // B fragment (32x16): col = l16 lane, elements 0..15 <-> K = k0 + 16*hf + 0..15 (contiguous)
    v16h bf = *(const v16h*)(Br + k0 + 16 * hf);
    acc = WMMA_F32_F16(af, bf, acc);
  }
#pragma unroll
  for (int v = 0; v < 8; v++) {
    int r = row0 + v + 8 * hf, c = col0 + l16;
    if (r < M && c < N) {
      float o = acc[v] + (bias ? bias[c] : 0.f);
      if (act) o = 0.5f * o * (1.f + erff(o * 0.70710678118f));   // exact GELU
      if (addend) o += addend[(size_t)r * ldc + c];
      C[(size_t)r * ldc + c] = o;
    }
  }
}

// ---------------- assemble x0 = [cls+pos ; LN2(patchout)+pos]  [B*197, 256] ----------------
__global__ __launch_bounds__(256) void assemble_x_kernel(
    const float* __restrict__ patchout, const float* __restrict__ g2, const float* __restrict__ b2,
    const float* __restrict__ cls, const float* __restrict__ pos, float* __restrict__ x) {
  __shared__ float sbuf[256];
  int row = blockIdx.x;  // b*197 + t
  int bi = row / vt::Tt, t = row % vt::Tt;
  int j = threadIdx.x;
  if (t == 0) {
    x[(size_t)row * vt::DIM + j] = cls[j] + pos[j];
    return;
  }
  float v    = patchout[((size_t)(bi * vt::NP + t - 1)) * vt::DIM + j];
  float mean = block_reduce_sum_256(v, sbuf) * (1.f / 256.f);
  float d    = v - mean;
  float var  = block_reduce_sum_256(d * d, sbuf) * (1.f / 256.f);
  float rinv = rsqrtf(var + 1e-5f);
  x[(size_t)row * vt::DIM + j] = d * rinv * g2[j] + b2[j] + pos[(size_t)t * vt::DIM + j];
}

// ---------------- LN + cast to f16 ----------------
__global__ __launch_bounds__(256) void ln_cast_kernel(
    const float* __restrict__ x, const float* __restrict__ g, const float* __restrict__ b,
    _Float16* __restrict__ y, int D) {
  __shared__ float sbuf[256];
  int row = blockIdx.x, tid = threadIdx.x;
  const float* xr = x + (size_t)row * D;
  int nit = (D + 255) / 256;
  float vals[4];
  float s = 0.f;
  for (int it = 0; it < nit; ++it) {
    int j = tid + it * 256;
    float v = (j < D) ? xr[j] : 0.f;
    vals[it] = v;
    s += v;
  }
  float mean = block_reduce_sum_256(s, sbuf) / (float)D;
  float vs = 0.f;
  for (int it = 0; it < nit; ++it) {
    int j = tid + it * 256;
    float d = (j < D) ? (vals[it] - mean) : 0.f;
    vs += d * d;
  }
  float var  = block_reduce_sum_256(vs, sbuf) / (float)D;
  float rinv = rsqrtf(var + 1e-5f);
  for (int it = 0; it < nit; ++it) {
    int j = tid + it * 256;
    if (j < D) y[(size_t)row * D + j] = (_Float16)((vals[it] - mean) * rinv * g[j] + b[j]);
  }
}

// ---------------- split abcde[B*T,1280] -> per-head f16 a,b,c,d [B,H,T,64] (c*=1/DH)
//                  and e transposed padded: et [B,H,64,TP] ----------------
__global__ void split_abcde_kernel(const float* __restrict__ abcde,
                                   _Float16* __restrict__ a, _Float16* __restrict__ b,
                                   _Float16* __restrict__ c, _Float16* __restrict__ d,
                                   _Float16* __restrict__ et, int n) {
  for (int idx = blockIdx.x * blockDim.x + threadIdx.x; idx < n; idx += gridDim.x * blockDim.x) {
    int row = idx / 1280, col = idx % 1280;
    int which = col >> 8, hc = col & 255;
    int h = hc >> 6, dh = hc & 63;
    int bi = row / vt::Tt, t = row % vt::Tt;
    float v = abcde[idx];
    if (which == 4) {
      et[((size_t)(bi * vt::H + h) * vt::DH + dh) * vt::TP + t] = (_Float16)v;
    } else {
      if (which == 2) v *= (1.f / 64.f);
      _Float16* dst = (which == 0) ? a : (which == 1) ? b : (which == 2) ? c : d;
      dst[((size_t)(bi * vt::H + h) * vt::Tt + t) * vt::DH + dh] = (_Float16)v;
    }
  }
}

// ---------------- fused trittention ----------------
__global__ __launch_bounds__(256) void trittention_kernel(
    const _Float16* __restrict__ ag, const _Float16* __restrict__ bg,
    const _Float16* __restrict__ cg, const _Float16* __restrict__ dg,
    const _Float16* __restrict__ et, float* __restrict__ z) {
  using namespace vt;
  __shared__ _Float16 sh_a[Tt * DH];
  __shared__ _Float16 sh_b[Tt * DH];
  __shared__ _Float16 sh_E[8][256];
  __shared__ float    sh_mz[8][16][2];
  __shared__ float    sh_m[16], sh_zz[16];
  __shared__ float    sh_acc[16 * 64];

  int tid = threadIdx.x;
  int w = tid >> 5, lane = tid & 31, l16 = lane & 15, hf = lane >> 4;
  int qt = blockIdx.x % NQT;
  int bh = blockIdx.x / NQT;

  {
    const unsigned int* pa = (const unsigned int*)(ag + (size_t)bh * Tt * DH);
    const unsigned int* pb = (const unsigned int*)(bg + (size_t)bh * Tt * DH);
    unsigned int* qa = (unsigned int*)sh_a;
    unsigned int* qb = (unsigned int*)sh_b;
    for (int i = tid; i < Tt * DH / 2; i += 256) { qa[i] = pa[i]; qb[i] = pb[i]; }
  }
  __syncthreads();

  int qrow = qt * 16 + l16;
  int qc = qrow < Tt ? qrow : Tt - 1;
  const _Float16* crow = cg + ((size_t)bh * Tt + qc) * DH;
  v16h cf0, cf1;
#pragma unroll
  for (int p = 0; p < 8; p++) {
    int kp = 2 * p + 16 * hf;
    cf0[2 * p] = crow[kp];      cf0[2 * p + 1] = crow[kp + 1];
    cf1[2 * p] = crow[kp + 32]; cf1[2 * p + 1] = crow[kp + 33];
  }

  int sbeg = w * SW, send = sbeg + SW;
  if (send > Tt) send = Tt;

  float m8[8], z8[8];
#pragma unroll
  for (int v = 0; v < 8; v++) { m8[v] = -1e30f; z8[v] = 0.f; }

  // ---- pass 1: per-lane online softmax stats ----
  for (int tt = 0; tt < NTT; ++tt) {
    int tb = tt * 16;
    bool colok = (tb + l16) < Tt;
    int tcl = colok ? tb + l16 : 0;
    const _Float16* brow = sh_b + tcl * DH + 16 * hf;
    v16h bv0 = *(const v16h*)brow;
    v16h bv1 = *(const v16h*)(brow + 32);
    for (int s = sbeg; s < send; ++s) {
      const _Float16* arow = sh_a + s * DH + 16 * hf;
      v16h av0 = *(const v16h*)arow;
      v16h av1 = *(const v16h*)(arow + 32);
      v16h g0 = av0 * bv0;     // G[d,t] = a_s[d]*b[t,d] in B-fragment order
      v16h g1 = av1 * bv1;
      v8f acc{};
      acc = WMMA_F32_F16(cf0, g0, acc);
      acc = WMMA_F32_F16(cf1, g1, acc);
      if (colok) {
#pragma unroll
        for (int v = 0; v < 8; v++) {
          float xx = acc[v];
          if (xx > m8[v]) { z8[v] = z8[v] * __expf(m8[v] - xx) + 1.f; m8[v] = xx; }
          else            { z8[v] += __expf(xx - m8[v]); }
        }
      }
    }
  }
#pragma unroll
  for (int off = 1; off < 16; off <<= 1) {
#pragma unroll
    for (int v = 0; v < 8; v++) {
      float mo = __shfl_xor(m8[v], off);
      float zo = __shfl_xor(z8[v], off);
      float mn = fmaxf(m8[v], mo);
      z8[v] = z8[v] * __expf(m8[v] - mn) + zo * __expf(mo - mn);
      m8[v] = mn;
    }
  }
  if (l16 == 0) {
#pragma unroll
    for (int v = 0; v < 8; v++) { sh_mz[w][hf * 8 + v][0] = m8[v]; sh_mz[w][hf * 8 + v][1] = z8[v]; }
  }
  __syncthreads();
  if (tid < 16) {
    float m = -1e30f, zz = 0.f;
    for (int ww = 0; ww < 8; ++ww) {
      float mo = sh_mz[ww][tid][0], zo = sh_mz[ww][tid][1];
      float mn = fmaxf(m, mo);
      zz = zz * __expf(m - mn) + zo * __expf(mo - mn);
      m = mn;
    }
    sh_m[tid] = m;
    sh_zz[tid] = zz;
  }
  __syncthreads();
  float mf[8];
#pragma unroll
  for (int v = 0; v < 8; v++) mf[v] = sh_m[hf * 8 + v];

  // ---- pass 2: numd (sum_r p * d_l) + nume (sum_l p * e_r) ----
  v8f zero8{};
  v8f accD[4];
#pragma unroll
  for (int nb = 0; nb < 4; nb++) accD[nb] = zero8;

  const _Float16* etb = et + (size_t)bh * DH * TP;

  for (int tt = 0; tt < NTT; ++tt) {
    int tb = tt * 16;
    bool colok = (tb + l16) < Tt;
    int tcl = colok ? tb + l16 : 0;
    const _Float16* brow = sh_b + tcl * DH + 16 * hf;
    v16h bv0 = *(const v16h*)brow;
    v16h bv1 = *(const v16h*)(brow + 32);
    // e B-fragments from transposed padded et: one contiguous v16h per dh-block
    v16h ef[4];
#pragma unroll
    for (int nb = 0; nb < 4; nb++) {
      int dh = nb * 16 + l16;
      ef[nb] = *(const v16h*)(etb + (size_t)dh * TP + tb + 16 * hf);
    }
    for (int s = sbeg; s < send; ++s) {
      const _Float16* arow = sh_a + s * DH + 16 * hf;
      v16h av0 = *(const v16h*)arow;
      v16h av1 = *(const v16h*)(arow + 32);
      v16h g0 = av0 * bv0;
      v16h g1 = av1 * bv1;
      v8f acc{};
      acc = WMMA_F32_F16(cf0, g0, acc);
      acc = WMMA_F32_F16(cf1, g1, acc);
      float Ev[8], rs[8];
#pragma unroll
      for (int v = 0; v < 8; v++) {
        float e = colok ? __expf(acc[v] - mf[v]) : 0.f;
        Ev[v] = e;
        rs[v] = e;
      }
#pragma unroll
      for (int off = 1; off < 16; off <<= 1)
#pragma unroll
        for (int v = 0; v < 8; v++) rs[v] += __shfl_xor(rs[v], off);
      const _Float16* drow = dg + ((size_t)bh * Tt + s) * DH + l16;
#pragma unroll
      for (int nb = 0; nb < 4; nb++) {
        float dv = (float)drow[nb * 16];
#pragma unroll
        for (int v = 0; v < 8; v++) accD[nb][v] += rs[v] * dv;
      }
      // relay E through per-wave LDS to re-layout as a WMMA A-fragment (K=16 real + 16 zero)
#pragma unroll
      for (int v = 0; v < 8; v++) sh_E[w][(v + 8 * hf) * 16 + l16] = (_Float16)Ev[v];
      asm volatile("s_wait_dscnt 0" ::: "memory");
      v16h Ef;
#pragma unroll
      for (int p = 0; p < 8; p++) {
        int kp = 2 * p + 16 * hf;
        Ef[2 * p]     = (kp     < 16) ? sh_E[w][l16 * 16 + kp]     : (_Float16)0.f;
        Ef[2 * p + 1] = (kp + 1 < 16) ? sh_E[w][l16 * 16 + kp + 1] : (_Float16)0.f;
      }
#pragma unroll
      for (int nb = 0; nb < 4; nb++) accD[nb] = WMMA_F32_F16(Ef, ef[nb], accD[nb]);
    }
  }

  for (int i = tid; i < 1024; i += 256) sh_acc[i] = 0.f;
  __syncthreads();
#pragma unroll
  for (int nb = 0; nb < 4; nb++)
#pragma unroll
    for (int v = 0; v < 8; v++)
      atomicAdd(&sh_acc[(v + 8 * hf) * 64 + nb * 16 + l16], accD[nb][v]);
  __syncthreads();
  int bi = bh / H, h = bh % H;
  for (int i = tid; i < 1024; i += 256) {
    int q = i >> 6, dh = i & 63;
    int qg = qt * 16 + q;
    if (qg < Tt)
      z[((size_t)(bi * Tt + qg)) * DIM + h * DH + dh] = sh_acc[i] / sh_zz[q];
  }
}

// ---------------- host driver ----------------
extern "C" void kernel_launch(void* const* d_in, const int* in_sizes, int n_in,
                              void* d_out, int out_size, void* d_ws, size_t ws_size,
                              hipStream_t stream) {
  using namespace vt;
  (void)in_sizes; (void)n_in; (void)out_size;

  const float* img       = (const float*)d_in[0];
  const float* p_ln1_g   = (const float*)d_in[1];
  const float* p_ln1_b   = (const float*)d_in[2];
  const float* patch_w   = (const float*)d_in[3];
  const float* patch_b   = (const float*)d_in[4];
  const float* p_ln2_g   = (const float*)d_in[5];
  const float* p_ln2_b   = (const float*)d_in[6];
  const float* pos_emb   = (const float*)d_in[7];
  const float* cls_tok   = (const float*)d_in[8];
  const float* attn_ln_g = (const float*)d_in[9];
  const float* attn_ln_b = (const float*)d_in[10];
  const float* w_abcde   = (const float*)d_in[11];
  const float* b_abcde   = (const float*)d_in[12];
  const float* w_out     = (const float*)d_in[13];
  const float* b_out     = (const float*)d_in[14];
  const float* ff_ln_g   = (const float*)d_in[15];
  const float* ff_ln_b   = (const float*)d_in[16];
  const float* ff_w1     = (const float*)d_in[17];
  const float* ff_b1     = (const float*)d_in[18];
  const float* ff_w2     = (const float*)d_in[19];
  const float* ff_b2     = (const float*)d_in[20];
  const float* fin_ln_g  = (const float*)d_in[21];
  const float* fin_ln_b  = (const float*)d_in[22];
  const float* head_w    = (const float*)d_in[23];
  const float* head_b    = (const float*)d_in[24];

  char* ws = (char*)d_ws;
  size_t off = 0;
  auto alloc = [&](size_t bytes) -> void* {
    void* p = ws + off;
    off += (bytes + 255) & ~size_t(255);
    return p;
  };
  const int ROWS = Bc * Tt;       // 394
  const int PROWS = Bc * NP;      // 392
  const int BH = Bc * H;          // 8

  _Float16* xpatch16 = (_Float16*)alloc((size_t)PROWS * PD * 2);
  _Float16* wtmp16   = (_Float16*)alloc((size_t)DIM * 1280 * 2);   // largest transposed weight
  float*    patchout = (float*)   alloc((size_t)PROWS * DIM * 4);
  float*    x        = (float*)   alloc((size_t)ROWS * DIM * 4);
  _Float16* xn16     = (_Float16*)alloc((size_t)ROWS * DIM * 2);
  float*    abcde    = (float*)   alloc((size_t)ROWS * 1280 * 4);
  _Float16* a16      = (_Float16*)alloc((size_t)BH * Tt * DH * 2);
  _Float16* b16      = (_Float16*)alloc((size_t)BH * Tt * DH * 2);
  _Float16* c16      = (_Float16*)alloc((size_t)BH * Tt * DH * 2);
  _Float16* d16      = (_Float16*)alloc((size_t)BH * Tt * DH * 2);
  _Float16* et16     = (_Float16*)alloc((size_t)BH * DH * TP * 2);
  float*    zbuf     = (float*)   alloc((size_t)ROWS * DIM * 4);
  _Float16* z16      = (_Float16*)alloc((size_t)ROWS * DIM * 2);
  float*    hbuf     = (float*)   alloc((size_t)ROWS * MLP * 4);
  _Float16* h16      = (_Float16*)alloc((size_t)ROWS * MLP * 2);
  if (off > ws_size) return;

  auto cgrid = [](int n) { int g = (n + 255) / 256; return g > 8192 ? 8192 : g; };
  auto gblocks = [](int M, int N) {
    int tiles = ((M + 15) / 16) * ((N + 15) / 16);
    return (tiles + 7) / 8;
  };

  patchify_ln_kernel<<<PROWS, 256, 0, stream>>>(img, p_ln1_g, p_ln1_b, xpatch16);
  cast_transpose_f16_kernel<<<cgrid(PD * DIM), 256, 0, stream>>>(patch_w, wtmp16, PD, DIM);
  gemm_wmma_kernel<<<gblocks(PROWS, DIM), 256, 0, stream>>>(
      xpatch16, PD, wtmp16, patch_b, nullptr, patchout, DIM, PROWS, DIM, PD, 0);
  assemble_x_kernel<<<ROWS, 256, 0, stream>>>(patchout, p_ln2_g, p_ln2_b, cls_tok, pos_emb, x);

  for (int i = 0; i < 2; i++) {
    // trittention block
    ln_cast_kernel<<<ROWS, 256, 0, stream>>>(x, attn_ln_g + i * DIM, attn_ln_b + i * DIM, xn16, DIM);
    cast_transpose_f16_kernel<<<cgrid(DIM * 1280), 256, 0, stream>>>(
        w_abcde + (size_t)i * DIM * 1280, wtmp16, DIM, 1280);
    gemm_wmma_kernel<<<gblocks(ROWS, 1280), 256, 0, stream>>>(
        xn16, DIM, wtmp16, b_abcde + i * 1280, nullptr, abcde, 1280, ROWS, 1280, DIM, 0);
    fill_zero_f16_kernel<<<cgrid(BH * DH * TP), 256, 0, stream>>>(et16, BH * DH * TP);
    split_abcde_kernel<<<cgrid(ROWS * 1280), 256, 0, stream>>>(abcde, a16, b16, c16, d16, et16,
                                                               ROWS * 1280);
    trittention_kernel<<<BH * NQT, 256, 0, stream>>>(a16, b16, c16, d16, et16, zbuf);
    cast_f32_f16_kernel<<<cgrid(ROWS * DIM), 256, 0, stream>>>(zbuf, z16, ROWS * DIM);
    cast_transpose_f16_kernel<<<cgrid(DIM * DIM), 256, 0, stream>>>(
        w_out + (size_t)i * DIM * DIM, wtmp16, DIM, DIM);
    gemm_wmma_kernel<<<gblocks(ROWS, DIM), 256, 0, stream>>>(
        z16, DIM, wtmp16, b_out + i * DIM, x, x, DIM, ROWS, DIM, DIM, 0);   // + residual

    // MLP block
    ln_cast_kernel<<<ROWS, 256, 0, stream>>>(x, ff_ln_g + i * DIM, ff_ln_b + i * DIM, xn16, DIM);
    cast_transpose_f16_kernel<<<cgrid(DIM * MLP), 256, 0, stream>>>(
        ff_w1 + (size_t)i * DIM * MLP, wtmp16, DIM, MLP);
    gemm_wmma_kernel<<<gblocks(ROWS, MLP), 256, 0, stream>>>(
        xn16, DIM, wtmp16, ff_b1 + i * MLP, nullptr, hbuf, MLP, ROWS, MLP, DIM, 1);  // GELU
    cast_f32_f16_kernel<<<cgrid(ROWS * MLP), 256, 0, stream>>>(hbuf, h16, ROWS * MLP);
    cast_transpose_f16_kernel<<<cgrid(MLP * DIM), 256, 0, stream>>>(
        ff_w2 + (size_t)i * MLP * DIM, wtmp16, MLP, DIM);
    gemm_wmma_kernel<<<gblocks(ROWS, DIM), 256, 0, stream>>>(
        h16, MLP, wtmp16, ff_b2 + i * DIM, x, x, DIM, ROWS, DIM, MLP, 0);   // + residual
  }

  // head: final LN, pick t=0 rows via lda = T*DIM
  ln_cast_kernel<<<ROWS, 256, 0, stream>>>(x, fin_ln_g, fin_ln_b, xn16, DIM);
  cast_transpose_f16_kernel<<<cgrid(DIM * NCLS), 256, 0, stream>>>(head_w, wtmp16, DIM, NCLS);
  gemm_wmma_kernel<<<gblocks(Bc, NCLS), 256, 0, stream>>>(
      xn16, Tt * DIM, wtmp16, head_b, nullptr, (float*)d_out, NCLS, Bc, NCLS, DIM, 0);
}
